// LocalAttention_57019985822499
// MI455X (gfx1250) — compile-verified, matrix operand order checked by reading, not run on hardware
//
#include <hip/hip_runtime.h>
#include <hip/hip_bf16.h>

#define D_MODEL 1024
#define SEQ     2048
#define BATCH   2
#define HEADS   16
#define HDIM    64
#define ROWS    (BATCH*SEQ)          /* 4096 */
#define WIN     128                  /* LOCAL_WINDOW/2 */
#define SCALE   0.125f               /* 1/sqrt(64) */
#define OUT_ELEMS ((size_t)ROWS * D_MODEL)             /* 4,194,304 */
#define ATTN_ELEMS ((size_t)BATCH * HEADS * SEQ * SEQ) /* 134,217,728 */

typedef __attribute__((ext_vector_type(16))) __bf16 v16bf;
typedef __attribute__((ext_vector_type(8)))  __bf16 v8bf;
typedef __attribute__((ext_vector_type(8)))  float  v8f;

// ---------------------------------------------------------------- helpers
__device__ __forceinline__ __bf16 f2bf(float f) {
  unsigned x = __float_as_uint(f);
  unsigned r = (x + 0x7FFFu + ((x >> 16) & 1u)) >> 16;   // RNE
  unsigned short s = (unsigned short)r;
  __bf16 o; __builtin_memcpy(&o, &s, 2); return o;
}

// 16x32 bf16 fragment, row-major source (also serves as B = rows of B^T).
// Lane L: row = L&15, half = L>>4; elements e<8 -> col 8h+e, e>=8 -> col 16+8h+(e-8)
__device__ __forceinline__ v16bf load_frag16x32(const __bf16* tile, int ld) {
  const int lane = threadIdx.x & 31;
  const int r = lane & 15, h = lane >> 4;
  const __bf16* p = tile + (size_t)r * ld + 8 * h;
  v8bf lo = *(const v8bf*)(p);
  v8bf hi = *(const v8bf*)(p + 16);
  v16bf f;
#pragma unroll
  for (int i = 0; i < 8; ++i) { f[i] = lo[i]; f[i + 8] = hi[i]; }
  return f;
}

// P fragment from fp32 LDS scores (convert to bf16 on the fly), ld = 288
__device__ __forceinline__ v16bf load_fragP(const float* sc, int kk) {
  const int lane = threadIdx.x & 31;
  const int r = lane & 15, h = lane >> 4;
  const float* p = sc + r * 288 + kk + 8 * h;
  v16bf f;
#pragma unroll
  for (int i = 0; i < 8; ++i) { f[i] = f2bf(p[i]); f[i + 8] = f2bf(p[16 + i]); }
  return f;
}

// B fragment for P@V from transposed V (vt[dim][key]); clamp keys to stay in-bounds
// (clamped region only multiplies zero-padded P columns).
__device__ __forceinline__ v16bf load_fragVt(const __bf16* vbase, int kbase) {
  const int lane = threadIdx.x & 31;
  const int n = lane & 15, h = lane >> 4;
  const __bf16* row = vbase + (size_t)n * SEQ;
  int c0 = kbase + 8 * h;      if (c0 > SEQ - 8) c0 = SEQ - 8;
  int c1 = kbase + 16 + 8 * h; if (c1 > SEQ - 8) c1 = SEQ - 8;
  v8bf lo = *(const v8bf*)(row + c0);
  v8bf hi = *(const v8bf*)(row + c1);
  v16bf f;
#pragma unroll
  for (int i = 0; i < 8; ++i) { f[i] = lo[i]; f[i + 8] = hi[i]; }
  return f;
}

// per-lane async global->LDS copy, 16 bytes (tracked by ASYNCcnt)
__device__ __forceinline__ void async_g2l_b128(unsigned lds_off, const void* gaddr) {
  asm volatile("global_load_async_to_lds_b128 %0, %1, off"
               :: "v"(lds_off), "v"(gaddr) : "memory");
}
__device__ __forceinline__ void wait_async0() {
  asm volatile("s_wait_asynccnt 0x0" ::: "memory");
}

// ---------------------------------------------------------------- prep: bf16 convert + weight transpose
__global__ __launch_bounds__(256) void la_prep(
    const float* __restrict__ x,
    const float* __restrict__ Wq, const float* __restrict__ Wk,
    const float* __restrict__ Wv, const float* __restrict__ Wo,
    __bf16* __restrict__ xb, __bf16* __restrict__ wt) {
  const size_t total = (size_t)ROWS * D_MODEL + 4ull * D_MODEL * D_MODEL;
  for (size_t i = (size_t)blockIdx.x * blockDim.x + threadIdx.x; i < total;
       i += (size_t)gridDim.x * blockDim.x) {
    if (i < (size_t)ROWS * D_MODEL) {
      xb[i] = f2bf(x[i]);
    } else {
      size_t j = i - (size_t)ROWS * D_MODEL;
      int mat = (int)(j / ((size_t)D_MODEL * D_MODEL));
      size_t e = j % ((size_t)D_MODEL * D_MODEL);
      int o  = (int)(e / D_MODEL);
      int in = (int)(e % D_MODEL);
      const float* W = (mat == 0) ? Wq : (mat == 1) ? Wk : (mat == 2) ? Wv : Wo;
      wt[j] = f2bf(W[(size_t)in * D_MODEL + o]);   // wt[mat][out][in] = W[in][out]
    }
  }
}

// ---------------------------------------------------------------- zero-fill attn (bandwidth floor)
__global__ __launch_bounds__(256) void la_fill_zero(float4* __restrict__ p, size_t n4) {
  for (size_t i = (size_t)blockIdx.x * blockDim.x + threadIdx.x; i < n4;
       i += (size_t)gridDim.x * blockDim.x)
    p[i] = make_float4(0.f, 0.f, 0.f, 0.f);
}

// ---------------------------------------------------------------- bf16 WMMA GEMM  C = A @ Bt^T + bias
// A: [M=ROWS][1024] row-major bf16, Bt: [N=1024][1024] row-major bf16 (pre-transposed).
// Block: 16 rows x 256 cols; A row-block staged once in LDS via async copy; 8 waves,
// each wave owns a 16x32 strip (2 accumulators share one A fragment per k-step).
// mode 0: out bf16 [B,H,S,HDIM] (Q/K)   mode 1: out bf16 [B,H,HDIM,S] (V transposed)
// mode 2: out fp32 row-major [ROWS][D_MODEL] (final projection)
__global__ __launch_bounds__(256) void la_wmma_gemm(
    const __bf16* __restrict__ A, const __bf16* __restrict__ Bt,
    const float* __restrict__ bias, void* __restrict__ out, int mode) {
  __shared__ __bf16 atile[16 * D_MODEL];   // 32 KB of the 320 KB WGP LDS

  const int wave = threadIdx.x >> 5;
  const int lane = threadIdx.x & 31;
  const int row0 = blockIdx.y * 16;
  const int col0 = blockIdx.x * 256 + wave * 32;

  // ---- stage A[row0..row0+15][0..1023] into LDS with async global->LDS b128
  {
    const unsigned base = (unsigned)(uintptr_t)&atile[0];
    for (int ci = threadIdx.x; ci < 16 * (D_MODEL / 8); ci += 256) {
      const int r  = ci >> 7;              // 128 16-byte chunks per row
      const int c8 = ci & 127;
      const __bf16* g = A + (size_t)(row0 + r) * D_MODEL + c8 * 8;
      const unsigned l = base + (unsigned)(r * D_MODEL + c8 * 8) * 2u;
      async_g2l_b128(l, g);
    }
    wait_async0();
  }
  __syncthreads();

  const __bf16* brow0 = Bt + (size_t)col0 * D_MODEL;
  const __bf16* brow1 = Bt + (size_t)(col0 + 16) * D_MODEL;

  v8f acc0 = {}, acc1 = {};
#pragma unroll 4
  for (int k = 0; k < D_MODEL; k += 32) {
    __builtin_prefetch(brow0 + (size_t)(lane & 15) * D_MODEL + k + 128, 0, 1);
    v16bf a  = load_frag16x32(atile + k, D_MODEL);     // ds_load_b128 x2
    v16bf b0 = load_frag16x32(brow0 + k, D_MODEL);
    v16bf b1 = load_frag16x32(brow1 + k, D_MODEL);
    acc0 = __builtin_amdgcn_wmma_f32_16x16x32_bf16(false, a, false, b0,
                                                   (short)0, acc0, false, false);
    acc1 = __builtin_amdgcn_wmma_f32_16x16x32_bf16(false, a, false, b1,
                                                   (short)0, acc1, false, false);
  }

  const int n  = lane & 15;
  const int mo = (lane >> 4) * 8;
#pragma unroll
  for (int ct = 0; ct < 2; ++ct) {
    const v8f acc = ct ? acc1 : acc0;
    const int col = col0 + ct * 16 + n;
    const float bv = bias ? bias[col] : 0.0f;
    if (mode == 2) {
      float* o = (float*)out;
#pragma unroll
      for (int t = 0; t < 8; ++t) {
        int r = row0 + mo + t;
        o[(size_t)r * D_MODEL + col] = acc[t] + bv;
      }
    } else {
      __bf16* o = (__bf16*)out;
      const int h = col >> 6, d = col & 63;
#pragma unroll
      for (int t = 0; t < 8; ++t) {
        int r = row0 + mo + t;
        int b = r >> 11, s = r & (SEQ - 1);
        size_t idx = (mode == 0)
          ? ((((size_t)b * HEADS + h) * SEQ) + s) * HDIM + d     // [B,H,S,Hd]
          : ((((size_t)b * HEADS + h) * HDIM) + d) * SEQ + s;    // [B,H,Hd,S]
        o[idx] = f2bf(acc[t] + bv);
      }
    }
  }
}

// ---------------------------------------------------------------- fused banded attention
// One block per (b*H+h, 16-query tile): WMMA scores -> LDS softmax -> attn scatter -> WMMA P@V
__global__ __launch_bounds__(256) void la_local_attn(
    const __bf16* __restrict__ qb, const __bf16* __restrict__ kb,
    const __bf16* __restrict__ vt, __bf16* __restrict__ ctx,
    float* __restrict__ attn) {
  __shared__ float sc[16][288];     // 16 queries x (<=272 keys, padded to 288)
  __shared__ float red[16][16];

  const int bh = blockIdx.x;                 // b*HEADS + h
  const int q0 = blockIdx.y * 16;
  const int kmin = (q0 - WIN > 0) ? (q0 - WIN) : 0;                  // 16-aligned
  const int kend = (q0 + 16 + WIN < SEQ) ? (q0 + 16 + WIN) : SEQ;    // 16-aligned
  const int ncol = kend - kmin;              // <= 272, multiple of 16
  const int nt   = ncol >> 4;

  const int wave = threadIdx.x >> 5;
  const int lane = threadIdx.x & 31;

  // zero LDS (pad columns must be 0 for the P@V phase)
  for (int i = threadIdx.x; i < 16 * 288; i += 256) (&sc[0][0])[i] = 0.0f;

  const __bf16* qtile = qb + ((size_t)bh * SEQ + q0) * HDIM;
  const __bf16* kbase = kb + ((size_t)bh * SEQ + kmin) * HDIM;
  v16bf a0 = load_frag16x32(qtile, HDIM);
  v16bf a1 = load_frag16x32(qtile + 32, HDIM);
  __syncthreads();

  // ---- phase 1: banded scores, one 16x16 key tile per wave round-robin
  const int n  = lane & 15;
  const int mo = (lane >> 4) * 8;
  for (int j = wave; j < nt; j += 8) {
    const __bf16* krow = kbase + (size_t)(j * 16) * HDIM;
    v16bf b0 = load_frag16x32(krow, HDIM);
    v16bf b1 = load_frag16x32(krow + 32, HDIM);
    v8f c = {};
    c = __builtin_amdgcn_wmma_f32_16x16x32_bf16(false, a0, false, b0, (short)0, c, false, false);
    c = __builtin_amdgcn_wmma_f32_16x16x32_bf16(false, a1, false, b1, (short)0, c, false, false);
#pragma unroll
    for (int t = 0; t < 8; ++t) {
      int qi = q0 + mo + t;
      int kj = kmin + j * 16 + n;
      int di = qi - kj; if (di < 0) di = -di;
      float s = c[t] * SCALE;
      if (di > WIN) s = -__builtin_inff();
      sc[mo + t][j * 16 + n] = s;
    }
  }
  __syncthreads();

  // ---- phase 2: row softmax + scatter band probabilities into dense attn
  const int row = threadIdx.x >> 4;
  const int t16 = threadIdx.x & 15;
  float m = -__builtin_inff();
  for (int c = t16; c < ncol; c += 16) m = fmaxf(m, sc[row][c]);
  red[row][t16] = m;
  __syncthreads();
  m = -__builtin_inff();
#pragma unroll
  for (int i = 0; i < 16; ++i) m = fmaxf(m, red[row][i]);
  float sum = 0.0f;
  for (int c = t16; c < ncol; c += 16) sum += __expf(sc[row][c] - m);
  __syncthreads();
  red[row][t16] = sum;
  __syncthreads();
  sum = 0.0f;
#pragma unroll
  for (int i = 0; i < 16; ++i) sum += red[row][i];
  const float inv = 1.0f / sum;
  float* arow = attn + ((size_t)bh * SEQ + (q0 + row)) * SEQ;
  for (int c = t16; c < ncol; c += 16) {
    float p = __expf(sc[row][c] - m) * inv;
    sc[row][c] = p;
    arow[kmin + c] = p;           // rest of the row is pre-zeroed
  }
  __syncthreads();

  // ---- phase 3: out = P (16 x <=288) @ V (keys x 64); waves 0..3 take 16 dims each
  if (wave < 4) {
    const int nd = wave * 16;
    const __bf16* vbase = vt + ((size_t)bh * HDIM + nd) * SEQ;
    v8f acc = {};
    const int kcap = (ncol + 31) & ~31;        // <= 288, pad cols of P are zero
    for (int kk = 0; kk < kcap; kk += 32) {
      v16bf a = load_fragP(&sc[0][0], kk);
      v16bf b = load_fragVt(vbase, kmin + kk);
      acc = __builtin_amdgcn_wmma_f32_16x16x32_bf16(false, a, false, b,
                                                    (short)0, acc, false, false);
    }
    const int b_ = bh >> 4, h_ = bh & 15;
#pragma unroll
    for (int t = 0; t < 8; ++t) {
      int s = q0 + mo + t;
      ctx[((size_t)(b_ * SEQ + s)) * D_MODEL + h_ * HDIM + nd + n] = f2bf(acc[t]);
    }
  }
}

// ---------------------------------------------------------------- launcher
extern "C" void kernel_launch(void* const* d_in, const int* in_sizes, int n_in,
                              void* d_out, int out_size, void* d_ws, size_t ws_size,
                              hipStream_t stream) {
  const float* x  = (const float*)d_in[0];
  const float* Wq = (const float*)d_in[1];
  const float* bq = (const float*)d_in[2];
  const float* Wk = (const float*)d_in[3];
  const float* bk = (const float*)d_in[4];
  const float* Wv = (const float*)d_in[5];
  const float* bv = (const float*)d_in[6];
  const float* Wo = (const float*)d_in[7];
  const float* bo = (const float*)d_in[8];

  float* out  = (float*)d_out;            // [B,S,D] fp32
  float* attn = out + OUT_ELEMS;          // [B,H,S,S] fp32

  // workspace carve-up (bf16), all 16B-aligned
  __bf16* wt  = (__bf16*)d_ws;                                  // 4 x [D][D] transposed weights
  __bf16* xb  = wt  + 4ull * D_MODEL * D_MODEL;                 // [ROWS][D]
  __bf16* qb  = xb  + OUT_ELEMS;                                // [B,H,S,Hd]
  __bf16* kb  = qb  + OUT_ELEMS;                                // [B,H,S,Hd]
  __bf16* vtp = kb  + OUT_ELEMS;                                // [B,H,Hd,S]
  __bf16* ctx = vtp + OUT_ELEMS;                                // [ROWS][D]

  la_prep<<<4096, 256, 0, stream>>>(x, Wq, Wk, Wv, Wo, xb, wt);
  la_fill_zero<<<32768, 256, 0, stream>>>((float4*)attn, ATTN_ELEMS / 4);

  dim3 ggrid(D_MODEL / 256, ROWS / 16);   // (4, 256)
  la_wmma_gemm<<<ggrid, 256, 0, stream>>>(xb, wt + 0ull * D_MODEL * D_MODEL, bq, qb,  0);
  la_wmma_gemm<<<ggrid, 256, 0, stream>>>(xb, wt + 1ull * D_MODEL * D_MODEL, bk, kb,  0);
  la_wmma_gemm<<<ggrid, 256, 0, stream>>>(xb, wt + 2ull * D_MODEL * D_MODEL, bv, vtp, 1);

  la_local_attn<<<dim3(BATCH * HEADS, SEQ / 16), 256, 0, stream>>>(qb, kb, vtp, ctx, attn);

  la_wmma_gemm<<<ggrid, 256, 0, stream>>>(ctx, wt + 3ull * D_MODEL * D_MODEL, bo, out, 2);
}
